// TriggerRec_37271726194993
// MI455X (gfx1250) — compile-verified, matrix operand order checked by reading, not run on hardware
//
#include <hip/hip_runtime.h>

// ---------------------------------------------------------------------------
// Problem constants (B=4, T=1024, E=1024, H=16, D=64)
// ---------------------------------------------------------------------------
#define Bc 4
#define Tc 1024
#define Ec 1024
#define Hc 16
#define Dc 64
#define ROWS (Bc * Tc)   // 4096

typedef __attribute__((ext_vector_type(16))) __bf16 v16bf;
typedef __attribute__((ext_vector_type(8)))  float  v8f;

union FragBF {
    unsigned int u32[8];
    v16bf bf;
};
union F8 {
    v8f v;
    float f[8];
};

__device__ __forceinline__ unsigned short f32_to_bf16(float f) {
    unsigned int u = __float_as_uint(f);
    u += 0x7FFFu + ((u >> 16) & 1u);   // round-to-nearest-even
    return (unsigned short)(u >> 16);
}

// LDS byte offset of a shared-memory pointer (generic AS ptr: low 32 bits)
__device__ __forceinline__ unsigned lds_offset(const void* p) {
    return (unsigned)(unsigned long long)p;
}

// CDNA5 async DMA: 16B per lane, global -> LDS, tracked by ASYNCcnt
__device__ __forceinline__ void async_copy_b128(unsigned lds_off, const void* gptr) {
    asm volatile("global_load_async_to_lds_b128 %0, %1, off"
                 :: "v"(lds_off), "v"((unsigned long long)gptr) : "memory");
}

// block of 256 threads: reduce two values, result broadcast to all threads
__device__ __forceinline__ void blockReduce2(float& a, float& b, float* s1, float* s2) {
    #pragma unroll
    for (int m = 16; m >= 1; m >>= 1) {
        a += __shfl_xor(a, m, 32);
        b += __shfl_xor(b, m, 32);
    }
    int w = threadIdx.x >> 5;
    if ((threadIdx.x & 31) == 0) { s1[w] = a; s2[w] = b; }
    __syncthreads();
    a = 0.f; b = 0.f;
    #pragma unroll
    for (int i = 0; i < 8; ++i) { a += s1[i]; b += s2[i]; }
    __syncthreads();
}

// ---------------------------------------------------------------------------
// Weight convert + transpose:  Wt[n][k] = bf16(W[k][n]),  E x E
// ---------------------------------------------------------------------------
__global__ void wconv_kernel(const float* __restrict__ W, unsigned short* __restrict__ Wt) {
    __shared__ unsigned short tile[32][33];
    int k0 = blockIdx.x * 32, n0 = blockIdx.y * 32;
    int tx = threadIdx.x, ty = threadIdx.y;           // 32 x 8
    #pragma unroll
    for (int i = 0; i < 4; ++i) {
        int kl = ty + i * 8;
        tile[kl][tx] = f32_to_bf16(W[(size_t)(k0 + kl) * Ec + n0 + tx]);
    }
    __syncthreads();
    #pragma unroll
    for (int i = 0; i < 4; ++i) {
        int nl = ty + i * 8;
        Wt[(size_t)(n0 + nl) * Ec + k0 + tx] = tile[tx][nl];
    }
}

// ---------------------------------------------------------------------------
// g = query_emb @ Wg + gamma ; bb = query_emb @ Wb + beta     (tiny: 4x1024)
// ---------------------------------------------------------------------------
__global__ void gb_kernel(const float* __restrict__ qe,
                          const float* __restrict__ Wg, const float* __restrict__ Wb,
                          const float* __restrict__ gamma, const float* __restrict__ beta,
                          float* __restrict__ g, float* __restrict__ bb) {
    int e = blockIdx.x * 256 + threadIdx.x;
    int b = blockIdx.y;
    float ag = gamma[e], ab = beta[e];
    for (int kk = 0; kk < Ec; ++kk) {
        float qv = qe[b * Ec + kk];
        ag += qv * Wg[(size_t)kk * Ec + e];
        ab += qv * Wb[(size_t)kk * Ec + e];
    }
    g[b * Ec + e]  = ag;
    bb[b * Ec + e] = ab;
}

// ---------------------------------------------------------------------------
// Conditional layernorm (ddof=1 std):  h = g*(x-mean)/(std+eps) + bb
// ---------------------------------------------------------------------------
__global__ void cln_kernel(const float* __restrict__ text,
                           const float* __restrict__ g, const float* __restrict__ bb,
                           float* __restrict__ hcln_f, unsigned short* __restrict__ hcln_bf) {
    __shared__ float s1[8], s2[8];
    int row = blockIdx.x;              // b*T + t
    int b = row >> 10;
    const float* x = text + (size_t)row * Ec;
    float vals[4];
    float s = 0.f, ss = 0.f;
    #pragma unroll
    for (int i = 0; i < 4; ++i) {
        float xv = x[threadIdx.x + i * 256];
        vals[i] = xv; s += xv; ss += xv * xv;
    }
    blockReduce2(s, ss, s1, s2);
    float mean = s * (1.f / 1024.f);
    float var  = fmaxf((ss - 1024.f * mean * mean) * (1.f / 1023.f), 0.f);
    float denom = sqrtf(var) + 1e-6f;
    #pragma unroll
    for (int i = 0; i < 4; ++i) {
        int e = threadIdx.x + i * 256;
        float hv = g[b * Ec + e] * (vals[i] - mean) / denom + bb[b * Ec + e];
        hcln_f [(size_t)row * Ec + e] = hv;
        hcln_bf[(size_t)row * Ec + e] = f32_to_bf16(hv);
    }
}

// ---------------------------------------------------------------------------
// Generic bf16 WMMA GEMM:  out = act( A[MxK] @ Bt[NxK]^T + bias + residual )
// block = 256 thr (8 waves), tile 64(M) x 128(N), K-step 32.
// Double-buffered LDS staging via CDNA5 async global->LDS DMA (ASYNCcnt).
// ---------------------------------------------------------------------------
__global__ void wmma_gemm_kernel(const unsigned short* __restrict__ A,
                                 const unsigned short* __restrict__ Bt,
                                 const float* __restrict__ bias,
                                 const float* __restrict__ residual,
                                 unsigned short* __restrict__ outBf,
                                 float* __restrict__ outF,
                                 int M, int N, int K, int act) {
    __shared__ unsigned short As[2][64][40];    // padded rows (40 ushorts)
    __shared__ unsigned short Bs[2][128][40];

    int tid  = threadIdx.x;
    int lane = tid & 31, wid = tid >> 5;
    int wr = wid & 3, wc = wid >> 2;
    int mb = blockIdx.y * 64, nb = blockIdx.x * 128;
    int lm = lane & 15, hf = lane >> 4;

    F8 acc[4];
    #pragma unroll
    for (int t = 0; t < 4; ++t)
        #pragma unroll
        for (int v = 0; v < 8; ++v) acc[t].f[v] = 0.f;

    int r = tid >> 2, c = (tid & 3) * 8;

    // per-thread async staging of one K-tile into LDS buffer `buf`
    auto stage = [&](int buf, int k0) {
        async_copy_b128(lds_offset(&As[buf][r][c]),
                        A  + (size_t)(mb + r) * K + k0 + c);
        async_copy_b128(lds_offset(&Bs[buf][r][c]),
                        Bt + (size_t)(nb + r) * K + k0 + c);
        async_copy_b128(lds_offset(&Bs[buf][r + 64][c]),
                        Bt + (size_t)(nb + r + 64) * K + k0 + c);
    };

    stage(0, 0);   // prime buffer 0

    for (int k0 = 0; k0 < K; k0 += 32) {
        int cur = (k0 >> 5) & 1;
        bool more = (k0 + 32) < K;
        if (more) stage(cur ^ 1, k0 + 32);   // prefetch next tile (other buffer)

        // wait until current tile's 3 transfers have landed (3 newer may remain)
        if (more) asm volatile("s_wait_asynccnt 3" ::: "memory");
        else      asm volatile("s_wait_asynccnt 0" ::: "memory");
        __syncthreads();

        FragBF a;
        const unsigned int* pa = (const unsigned int*)&As[cur][wr * 16 + lm][0];
        #pragma unroll
        for (int i = 0; i < 4; ++i) {
            a.u32[i]     = pa[hf * 4 + i];
            a.u32[4 + i] = pa[8 + hf * 4 + i];
        }
        #pragma unroll
        for (int t = 0; t < 4; ++t) {
            FragBF bf;
            const unsigned int* pb = (const unsigned int*)&Bs[cur][wc * 64 + t * 16 + lm][0];
            #pragma unroll
            for (int i = 0; i < 4; ++i) {
                bf.u32[i]     = pb[hf * 4 + i];
                bf.u32[4 + i] = pb[8 + hf * 4 + i];
            }
            acc[t].v = __builtin_amdgcn_wmma_f32_16x16x32_bf16(
                false, a.bf, false, bf.bf, (short)0, acc[t].v, false, false);
        }
        __syncthreads();   // all waves done reading `cur` before it is re-staged
    }

    #pragma unroll
    for (int t = 0; t < 4; ++t) {
        int n = nb + wc * 64 + t * 16 + lm;
        float bv = bias ? bias[n] : 0.f;
        #pragma unroll
        for (int v = 0; v < 8; ++v) {
            int m = mb + wr * 16 + hf * 8 + v;
            size_t idx = (size_t)m * N + n;
            float val = acc[t].f[v] + bv;
            if (residual) val += residual[idx];
            if (act == 1) val = 0.5f * val * (1.f + erff(val * 0.70710678118654752f));
            if (outF)  outF[idx]  = val;
            if (outBf) outBf[idx] = f32_to_bf16(val);
        }
    }
}

// ---------------------------------------------------------------------------
// V transpose:  vt[b][h][d][t] = v[b*T+t][h*64+d]   (bf16)
// ---------------------------------------------------------------------------
__global__ void vt_kernel(const unsigned short* __restrict__ v, unsigned short* __restrict__ vt) {
    __shared__ unsigned short tile[32][33];
    int bh = blockIdx.z; int b = bh >> 4, h = bh & 15;
    int t0 = blockIdx.x * 32, d0 = blockIdx.y * 32;
    int tx = threadIdx.x, ty = threadIdx.y;   // 32 x 8
    #pragma unroll
    for (int i = 0; i < 4; ++i) {
        int tl = ty + i * 8;
        tile[tl][tx] = v[(size_t)(b * Tc + t0 + tl) * Ec + h * Dc + d0 + tx];
    }
    __syncthreads();
    #pragma unroll
    for (int i = 0; i < 4; ++i) {
        int dl = ty + i * 8;
        vt[(size_t)(bh * Dc + d0 + dl) * Tc + t0 + tx] = tile[tx][dl];
    }
}

// ---------------------------------------------------------------------------
// Flash attention:  per wave = 16 query rows, full online-softmax over T=1024
// grid = (T/128, B*H), block = 256 (8 waves)
// ---------------------------------------------------------------------------
__global__ void attn_kernel(const unsigned short* __restrict__ q,
                            const unsigned short* __restrict__ k,
                            const unsigned short* __restrict__ vt,
                            const int* __restrict__ mask,
                            unsigned short* __restrict__ attn) {
    __shared__ unsigned short Pl[8][16][34];   // per-wave P staging (C->A layout)

    int tid = threadIdx.x, lane = tid & 31, wid = tid >> 5;
    int lm = lane & 15, hf = lane >> 4;
    int bh = blockIdx.y; int b = bh >> 4, h = bh & 15;
    int qw = blockIdx.x * 128 + wid * 16;

    // Q fragments (16 rows x 64 d) held in registers for the whole pass
    FragBF a0, a1;
    {
        const unsigned int* p =
            (const unsigned int*)(q + (size_t)(b * Tc + qw + lm) * Ec + h * Dc);
        #pragma unroll
        for (int i = 0; i < 4; ++i) {
            a0.u32[i]     = p[hf * 4 + i];
            a0.u32[4 + i] = p[8 + hf * 4 + i];
            a1.u32[i]     = p[16 + hf * 4 + i];
            a1.u32[4 + i] = p[24 + hf * 4 + i];
        }
    }

    F8 O[4];
    float Mr[8], Lr[8];
    #pragma unroll
    for (int v = 0; v < 8; ++v) { Mr[v] = -3.0e38f; Lr[v] = 0.f; }
    #pragma unroll
    for (int dt = 0; dt < 4; ++dt)
        #pragma unroll
        for (int v = 0; v < 8; ++v) O[dt].f[v] = 0.f;

    for (int kt = 0; kt < Tc; kt += 32) {
        // ---- S = (Q K^T) / 8  + mask bias  (two 16x16 tiles = 32 keys) ----
        F8 S[2];
        #pragma unroll
        for (int s = 0; s < 2; ++s) {
            FragBF kb0, kb1;
            int key = kt + s * 16 + lm;
            const unsigned int* p =
                (const unsigned int*)(k + (size_t)(b * Tc + key) * Ec + h * Dc);
            #pragma unroll
            for (int i = 0; i < 4; ++i) {
                kb0.u32[i]     = p[hf * 4 + i];
                kb0.u32[4 + i] = p[8 + hf * 4 + i];
                kb1.u32[i]     = p[16 + hf * 4 + i];
                kb1.u32[4 + i] = p[24 + hf * 4 + i];
            }
            F8 z;
            #pragma unroll
            for (int v = 0; v < 8; ++v) z.f[v] = 0.f;
            v8f tmp = __builtin_amdgcn_wmma_f32_16x16x32_bf16(
                false, a0.bf, false, kb0.bf, (short)0, z.v, false, false);
            S[s].v = __builtin_amdgcn_wmma_f32_16x16x32_bf16(
                false, a1.bf, false, kb1.bf, (short)0, tmp, false, false);
            float mb_ = (1.f - (float)mask[b * Tc + key]) * -10000.f;
            #pragma unroll
            for (int v = 0; v < 8; ++v) S[s].f[v] = S[s].f[v] * 0.125f + mb_;
        }

        // ---- online softmax (row = vgpr index; cols across 16-lane half) ----
        #pragma unroll
        for (int v = 0; v < 8; ++v) {
            float t = fmaxf(S[0].f[v], S[1].f[v]);
            #pragma unroll
            for (int m = 8; m >= 1; m >>= 1) t = fmaxf(t, __shfl_xor(t, m, 32));
            float nM   = fmaxf(Mr[v], t);
            float corr = __expf(Mr[v] - nM);
            Mr[v] = nM;
            S[0].f[v] = __expf(S[0].f[v] - nM);
            S[1].f[v] = __expf(S[1].f[v] - nM);
            float rs = S[0].f[v] + S[1].f[v];
            #pragma unroll
            for (int m = 8; m >= 1; m >>= 1) rs += __shfl_xor(rs, m, 32);
            Lr[v] = Lr[v] * corr + rs;
            #pragma unroll
            for (int dt = 0; dt < 4; ++dt) O[dt].f[v] *= corr;
        }

        // ---- C-layout P -> LDS -> A-layout fragment (wave-private) ----
        #pragma unroll
        for (int v = 0; v < 8; ++v) {
            Pl[wid][hf * 8 + v][lm]      = f32_to_bf16(S[0].f[v]);
            Pl[wid][hf * 8 + v][16 + lm] = f32_to_bf16(S[1].f[v]);
        }
        FragBF pfr;
        const unsigned int* pp = (const unsigned int*)&Pl[wid][lm][0];
        #pragma unroll
        for (int i = 0; i < 4; ++i) {
            pfr.u32[i]     = pp[hf * 4 + i];
            pfr.u32[4 + i] = pp[8 + hf * 4 + i];
        }

        // ---- O += P @ V  (V pre-transposed: contiguous over keys) ----
        #pragma unroll
        for (int dt = 0; dt < 4; ++dt) {
            FragBF vb;
            const unsigned int* pv =
                (const unsigned int*)(vt + (size_t)(bh * Dc + dt * 16 + lm) * Tc + kt);
            #pragma unroll
            for (int i = 0; i < 4; ++i) {
                vb.u32[i]     = pv[hf * 4 + i];
                vb.u32[4 + i] = pv[8 + hf * 4 + i];
            }
            O[dt].v = __builtin_amdgcn_wmma_f32_16x16x32_bf16(
                false, pfr.bf, false, vb.bf, (short)0, O[dt].v, false, false);
        }
    }

    // ---- epilogue: O / L -> attn (bf16, [b*T+t][h*64+d]) ----
    #pragma unroll
    for (int v = 0; v < 8; ++v) {
        float inv = 1.f / Lr[v];
        int trow = qw + hf * 8 + v;
        #pragma unroll
        for (int dt = 0; dt < 4; ++dt)
            attn[(size_t)(b * Tc + trow) * Ec + h * Dc + dt * 16 + lm] =
                f32_to_bf16(O[dt].f[v] * inv);
    }
}

// ---------------------------------------------------------------------------
// Post layernorm (ddof=0): y = (x-mu)/sqrt(var+1e-5)*ln_g + ln_b  -> bf16
// ---------------------------------------------------------------------------
__global__ void ln_kernel(const float* __restrict__ x,
                          const float* __restrict__ ln_g, const float* __restrict__ ln_b,
                          unsigned short* __restrict__ y_bf) {
    __shared__ float s1[8], s2[8];
    int row = blockIdx.x;
    const float* xr = x + (size_t)row * Ec;
    float vals[4];
    float s = 0.f, ss = 0.f;
    #pragma unroll
    for (int i = 0; i < 4; ++i) {
        float xv = xr[threadIdx.x + i * 256];
        vals[i] = xv; s += xv; ss += xv * xv;
    }
    blockReduce2(s, ss, s1, s2);
    float mu  = s * (1.f / 1024.f);
    float var = fmaxf(ss * (1.f / 1024.f) - mu * mu, 0.f);
    float rstd = rsqrtf(var + 1e-5f);
    #pragma unroll
    for (int i = 0; i < 4; ++i) {
        int e = threadIdx.x + i * 256;
        float yv = (vals[i] - mu) * rstd * ln_g[e] + ln_b[e];
        y_bf[(size_t)row * Ec + e] = f32_to_bf16(yv);
    }
}

// ---------------------------------------------------------------------------
// Heads: p_s = sigmoid(x@w_head + b_head), p_e = sigmoid(x@w_tail + b_tail)
// ---------------------------------------------------------------------------
__global__ void head_kernel(const float* __restrict__ x2,
                            const float* __restrict__ wh, const float* __restrict__ bh_,
                            const float* __restrict__ wt, const float* __restrict__ bt_,
                            float* __restrict__ out) {
    __shared__ float s1[8], s2[8];
    int row = blockIdx.x;
    const float* xr = x2 + (size_t)row * Ec;
    float a = 0.f, b = 0.f;
    for (int i = threadIdx.x; i < Ec; i += 256) {
        float xv = xr[i];
        a += xv * wh[i];
        b += xv * wt[i];
    }
    blockReduce2(a, b, s1, s2);
    if (threadIdx.x == 0) {
        out[row]         = 1.f / (1.f + __expf(-(a + bh_[0])));
        out[ROWS + row]  = 1.f / (1.f + __expf(-(b + bt_[0])));
    }
}

// ---------------------------------------------------------------------------
// Host-side orchestration
// ---------------------------------------------------------------------------
extern "C" void kernel_launch(void* const* d_in, const int* in_sizes, int n_in,
                              void* d_out, int out_size, void* d_ws, size_t ws_size,
                              hipStream_t stream) {
    (void)in_sizes; (void)n_in; (void)out_size; (void)ws_size;

    const float* query_emb = (const float*)d_in[0];
    const float* text_emb  = (const float*)d_in[1];
    const int*   mask      = (const int*)  d_in[2];
    const float* Wg   = (const float*)d_in[3];
    const float* Wb   = (const float*)d_in[4];
    const float* gamma= (const float*)d_in[5];
    const float* beta = (const float*)d_in[6];
    const float* Wq   = (const float*)d_in[7];
    const float* bq   = (const float*)d_in[8];
    const float* Wk   = (const float*)d_in[9];
    const float* bk   = (const float*)d_in[10];
    const float* Wv   = (const float*)d_in[11];
    const float* bv   = (const float*)d_in[12];
    const float* Wo   = (const float*)d_in[13];
    const float* bo   = (const float*)d_in[14];
    const float* ln_g = (const float*)d_in[15];
    const float* ln_b = (const float*)d_in[16];
    const float* Wh   = (const float*)d_in[17];
    const float* bh   = (const float*)d_in[18];
    const float* w_head = (const float*)d_in[19];
    const float* b_head = (const float*)d_in[20];
    const float* w_tail = (const float*)d_in[21];
    const float* b_tail = (const float*)d_in[22];

    float* OUT = (float*)d_out;
    float* p_out   = OUT;              // p_s [4096] then p_e [4096]
    float* hcln_f  = OUT + 2 * ROWS;   // h_cln f32 output (also residual)

    // workspace layout (bytes)
    const size_t MB = 1u << 20;
    char* ws = (char*)d_ws;
    unsigned short* WqT   = (unsigned short*)(ws + 0 * MB);
    unsigned short* WkT   = (unsigned short*)(ws + 2 * MB);
    unsigned short* WvT   = (unsigned short*)(ws + 4 * MB);
    unsigned short* WoT   = (unsigned short*)(ws + 6 * MB);
    unsigned short* WhT   = (unsigned short*)(ws + 8 * MB);
    unsigned short* HCb   = (unsigned short*)(ws + 10 * MB);  // h_cln bf16
    unsigned short* Qb    = (unsigned short*)(ws + 18 * MB);
    unsigned short* Kb    = (unsigned short*)(ws + 26 * MB);
    unsigned short* Vb    = (unsigned short*)(ws + 34 * MB);
    unsigned short* VTb   = (unsigned short*)(ws + 42 * MB);  // V^T [b,h,d,t]
    unsigned short* ATb   = (unsigned short*)(ws + 50 * MB);  // attn bf16
    float*          Xf    = (float*)        (ws + 58 * MB);   // post-Wo f32
    unsigned short* XLNb  = (unsigned short*)(ws + 74 * MB);  // LN(x) bf16
    float*          X2f   = (float*)        (ws + 82 * MB);   // gelu out f32
    float*          Gf    = (float*)        (ws + 98 * MB);
    float*          BBf   = (float*)        (ws + 98 * MB + 16384);

    dim3 tblk(32, 8);
    dim3 tgrd(Ec / 32, Ec / 32);
    // weight convert+transpose to bf16 [N][K]
    wconv_kernel<<<tgrd, tblk, 0, stream>>>(Wq, WqT);
    wconv_kernel<<<tgrd, tblk, 0, stream>>>(Wk, WkT);
    wconv_kernel<<<tgrd, tblk, 0, stream>>>(Wv, WvT);
    wconv_kernel<<<tgrd, tblk, 0, stream>>>(Wo, WoT);
    wconv_kernel<<<tgrd, tblk, 0, stream>>>(Wh, WhT);

    // conditional scale/shift (tiny f32 GEMMs)
    gb_kernel<<<dim3(Ec / 256, Bc), 256, 0, stream>>>(query_emb, Wg, Wb, gamma, beta, Gf, BBf);

    // conditional layernorm -> h_cln (f32 output + bf16 activation)
    cln_kernel<<<ROWS, 256, 0, stream>>>(text_emb, Gf, BBf, hcln_f, HCb);

    // QKV projections (bf16 WMMA, async-DMA staged)
    dim3 ggrd(Ec / 128, ROWS / 64);
    wmma_gemm_kernel<<<ggrd, 256, 0, stream>>>(HCb, WqT, bq, nullptr, Qb, nullptr, ROWS, Ec, Ec, 0);
    wmma_gemm_kernel<<<ggrd, 256, 0, stream>>>(HCb, WkT, bk, nullptr, Kb, nullptr, ROWS, Ec, Ec, 0);
    wmma_gemm_kernel<<<ggrd, 256, 0, stream>>>(HCb, WvT, bv, nullptr, Vb, nullptr, ROWS, Ec, Ec, 0);

    // V transpose for contiguous PV fragments
    vt_kernel<<<dim3(Tc / 32, Dc / 32, Bc * Hc), tblk, 0, stream>>>(Vb, VTb);

    // flash attention
    attn_kernel<<<dim3(Tc / 128, Bc * Hc), 256, 0, stream>>>(Qb, Kb, VTb, mask, ATb);

    // output projection + residual (f32 out)
    wmma_gemm_kernel<<<ggrd, 256, 0, stream>>>(ATb, WoT, bo, hcln_f, nullptr, Xf, ROWS, Ec, Ec, 0);

    // layernorm -> bf16
    ln_kernel<<<ROWS, 256, 0, stream>>>(Xf, ln_g, ln_b, XLNb);

    // hidden GEMM + GeLU (f32 out)
    wmma_gemm_kernel<<<ggrd, 256, 0, stream>>>(XLNb, WhT, bh, nullptr, nullptr, X2f, ROWS, Ec, Ec, 1);

    // sigmoid heads
    head_kernel<<<ROWS, 256, 0, stream>>>(X2f, w_head, b_head, w_tail, b_tail, p_out);
}
